// Model_GRU_ATT_GRU_48026324304000
// MI455X (gfx1250) — matrix operand order repaired and checked by reading, not compiled, not run on hardware
//
#include <hip/hip_runtime.h>
#include <hip/hip_bf16.h>

// ---------------------------------------------------------------------------
// DIEN (GRU -> DIN attention -> GRU -> FCN) for MI455X / gfx1250, wave32.
// All GEMMs use v_wmma_f32_16x16x32_f16 (f16 in, f32 accumulate).
// Attention weights staged to LDS via global_load_async_to_lds_b128.
// ---------------------------------------------------------------------------

typedef __attribute__((ext_vector_type(16))) _Float16 v16h;
typedef __attribute__((ext_vector_type(8)))  float    v8f;

#define B_SZ 4096
#define T_SZ 200
#define E_SZ 64
#define H_SZ 64

#define WMMA_F16(a, b, c) \
  __builtin_amdgcn_wmma_f32_16x16x32_f16(false, (a), false, (b), (short)0, (c), false, false)

// ---------------- fragment helpers (wave32, 16x16x32 f16) ------------------
// A (16x32, 16-bit): lane l -> row m=l&15; lanes 0-15 hold K 0..7/16..23,
// lanes 16-31 hold K 8..15/24..31, 2 halves per VGPR.
__device__ __forceinline__ v16h load_a_f16(const _Float16* base, int stride, int k0) {
  const int l = threadIdx.x & 31, m = l & 15, hi = (l >> 4) * 8;
  const _Float16* p = base + m * stride + k0;
  v16h a;
#pragma unroll
  for (int j = 0; j < 8; ++j) {
    int k = ((j < 4) ? 0 : 16) + hi + 2 * (j & 3);
    a[2 * j]     = p[k];
    a[2 * j + 1] = p[k + 1];
  }
  return a;
}

// A fragment built from f32 LDS state (on-the-fly f32->f16)
__device__ __forceinline__ v16h make_a_f32(const float* S, int stride, int k0) {
  const int l = threadIdx.x & 31, m = l & 15, hi = (l >> 4) * 8;
  const float* p = S + m * stride + k0;
  v16h a;
#pragma unroll
  for (int j = 0; j < 8; ++j) {
    int k = ((j < 4) ? 0 : 16) + hi + 2 * (j & 3);
    a[2 * j]     = (_Float16)p[k];
    a[2 * j + 1] = (_Float16)p[k + 1];
  }
  return a;
}

// A fragment of (r * h): r from rz[.,0..63] (stride 128), h from h[.,0..63]
__device__ __forceinline__ v16h make_a_rh(const float* rz, const float* h, int k0) {
  const int l = threadIdx.x & 31, m = l & 15, hi = (l >> 4) * 8;
  v16h a;
#pragma unroll
  for (int j = 0; j < 8; ++j) {
    int k = k0 + ((j < 4) ? 0 : 16) + hi + 2 * (j & 3);
    a[2 * j]     = (_Float16)(rz[m * 128 + k]     * h[m * 64 + k]);
    a[2 * j + 1] = (_Float16)(rz[m * 128 + k + 1] * h[m * 64 + k + 1]);
  }
  return a;
}

// B (32x16, 16-bit): lane l -> col n=l&15; lanes 0-15 K 0..7/16..23, lanes 16-31 K 8..15/24..31
__device__ __forceinline__ v16h load_b_f16(const _Float16* Bm, int stride, int k0, int n0) {
  const int l = threadIdx.x & 31, n = l & 15, kb = (l >> 4) * 8;
  const _Float16* p = Bm + (size_t)(k0 + kb) * stride + n0 + n;
  v16h b;
#pragma unroll
  for (int j = 0; j < 8; ++j) {
    b[2 * j]     = p[(((j < 4) ? 0 : 8) + 2 * j) * stride];
    b[2 * j + 1] = p[(((j < 4) ? 0 : 8) + 2 * j + 1) * stride];
  }
  return b;
}

// C/D (16x16 f32): lane l -> col n=l&15, rows m0..m0+7 with m0=(l>>4)*8
__device__ __forceinline__ v8f bias_frag(const float* bias, int n0) {
  float bv = bias[n0 + (threadIdx.x & 15)];
  v8f c;
#pragma unroll
  for (int r = 0; r < 8; ++r) c[r] = bv;
  return c;
}

__device__ __forceinline__ v8f bias_frag_lim(const float* bias, int n0, int lim) {
  int n = n0 + (threadIdx.x & 15);
  float bv = (n < lim) ? bias[n] : 0.f;
  v8f c;
#pragma unroll
  for (int r = 0; r < 8; ++r) c[r] = bv;
  return c;
}

__device__ __forceinline__ void store_d_f32(float* S, int stride, int n0, v8f d) {
  const int l = threadIdx.x & 31, n = l & 15, m0 = (l >> 4) * 8;
#pragma unroll
  for (int r = 0; r < 8; ++r) S[(m0 + r) * stride + n0 + n] = d[r];
}

__device__ __forceinline__ void store_d_f16(_Float16* S, int stride, int n0, v8f d) {
  const int l = threadIdx.x & 31, n = l & 15, m0 = (l >> 4) * 8;
#pragma unroll
  for (int r = 0; r < 8; ++r) S[(m0 + r) * stride + n0 + n] = (_Float16)d[r];
}

__device__ __forceinline__ float sigm(float x) { return 1.f / (1.f + __expf(-x)); }

// ---------------------------------------------------------------------------
// Kernel: f32 -> f16 weight conversion with zero padding to WMMA-friendly dims
// ---------------------------------------------------------------------------
__global__ void k_cvt_pad(const float* __restrict__ src, _Float16* __restrict__ dst,
                          int R, int C, int R2, int C2) {
  int i = blockIdx.x * blockDim.x + threadIdx.x;
  int n = R2 * C2;
  if (i >= n) return;
  int r = i / C2, c = i % C2;
  dst[i] = (r < R && c < C) ? (_Float16)src[r * C + c] : (_Float16)0.f;
}

// ---------------------------------------------------------------------------
// Kernel: embedding gathers + hist_sum (tables are L2-resident: 25.6 MB)
// ---------------------------------------------------------------------------
__global__ void k_gather(const int* __restrict__ u, const int* __restrict__ iid,
                         const int* __restrict__ hist,
                         const float* __restrict__ item_w, const float* __restrict__ user_w,
                         _Float16* __restrict__ hist_f16, float* __restrict__ hsum,
                         float* __restrict__ item_f32, _Float16* __restrict__ item_f16,
                         float* __restrict__ user_f32) {
  __shared__ float part[4][64];
  const int b = blockIdx.x;
  const int tid = threadIdx.x;
  if (tid < 64) {
    float uv = user_w[(size_t)u[b] * 64 + tid];
    float iv = item_w[(size_t)iid[b] * 64 + tid];
    user_f32[(size_t)b * 64 + tid] = uv;
    item_f32[(size_t)b * 64 + tid] = iv;
    item_f16[(size_t)b * 64 + tid] = (_Float16)iv;
  }
  const int e = tid & 63, chunk = tid >> 6;  // 4 chunks of timesteps
  float acc = 0.f;
  for (int t = chunk; t < T_SZ; t += 4) {
    float v = item_w[(size_t)hist[(size_t)b * T_SZ + t] * 64 + e];
    hist_f16[((size_t)b * T_SZ + t) * 64 + e] = (_Float16)v;
    acc += v;
  }
  part[chunk][e] = acc;
  __syncthreads();
  if (tid < 64) hsum[(size_t)b * 64 + tid] = part[0][tid] + part[1][tid] + part[2][tid] + part[3][tid];
}

// ---------------------------------------------------------------------------
// Kernel: fused GRU scan. One wave owns 16 batch rows; per step the input
// projection x@W and recurrent h@U accumulate into the same WMMA C tile.
// Weight B-fragments are loop-invariant -> compiler keeps them in VGPRs
// across all 200 steps (verified in round-1 disasm).
// ---------------------------------------------------------------------------
__global__ void __launch_bounds__(128) k_gru_scan(
    const _Float16* __restrict__ xseq,                      // [B,T,64] f16
    const _Float16* __restrict__ Wg, const _Float16* __restrict__ Ug,  // [64,128]
    const _Float16* __restrict__ Wc, const _Float16* __restrict__ Uc,  // [64,64]
    const float* __restrict__ bg, const float* __restrict__ bc,
    const float* __restrict__ mask,                         // [B,T]
    _Float16* __restrict__ seq_out,                         // [B,T,64] f16 or null
    float* __restrict__ final_out) {                        // [B,64] f32 or null
  __shared__ float sh_h[4][16][64];    // 16 KB
  __shared__ float sh_rz[4][16][128];  // 32 KB
  const int wave = threadIdx.x >> 5;
  const int lane = threadIdx.x & 31;
  const int r0 = (blockIdx.x * 4 + wave) * 16;  // first batch row of this wave
  float* hw  = &sh_h[wave][0][0];
  float* rzw = &sh_rz[wave][0][0];

  for (int i = lane; i < 16 * 64; i += 32) hw[i] = 0.f;
  __syncthreads();

  const int n = lane & 15, m0 = (lane >> 4) * 8;

  for (int t = 0; t < T_SZ; ++t) {
    const _Float16* xrow = xseq + ((size_t)r0 * T_SZ + t) * 64;
    if (t + 1 < T_SZ)  // prefetch next step's x tile (global_prefetch_b8)
      __builtin_prefetch(xrow + 64 + (lane & 15) * T_SZ * 64, 0, 0);
    v16h ax0 = load_a_f16(xrow, T_SZ * 64, 0);
    v16h ax1 = load_a_f16(xrow, T_SZ * 64, 32);
    v16h ah0 = make_a_f32(hw, 64, 0);
    v16h ah1 = make_a_f32(hw, 64, 32);

    // gates: [r|z] = sigmoid(x@Wg + h@Ug + bg), 8 N-tiles of 16
#pragma unroll
    for (int tn = 0; tn < 8; ++tn) {
      v8f c = bias_frag(bg, tn * 16);
      c = WMMA_F16(ax0, load_b_f16(Wg, 128, 0,  tn * 16), c);
      c = WMMA_F16(ax1, load_b_f16(Wg, 128, 32, tn * 16), c);
      c = WMMA_F16(ah0, load_b_f16(Ug, 128, 0,  tn * 16), c);
      c = WMMA_F16(ah1, load_b_f16(Ug, 128, 32, tn * 16), c);
#pragma unroll
      for (int r = 0; r < 8; ++r) c[r] = sigm(c[r]);
      store_d_f32(rzw, 128, tn * 16, c);
    }
    __syncthreads();

    v16h ar0 = make_a_rh(rzw, hw, 0);
    v16h ar1 = make_a_rh(rzw, hw, 32);

    // candidate + state update fused per 16-col tile
#pragma unroll
    for (int tn = 0; tn < 4; ++tn) {
      v8f c = bias_frag(bc, tn * 16);
      c = WMMA_F16(ax0, load_b_f16(Wc, 64, 0,  tn * 16), c);
      c = WMMA_F16(ax1, load_b_f16(Wc, 64, 32, tn * 16), c);
      c = WMMA_F16(ar0, load_b_f16(Uc, 64, 0,  tn * 16), c);
      c = WMMA_F16(ar1, load_b_f16(Uc, 64, 32, tn * 16), c);
#pragma unroll
      for (int r = 0; r < 8; ++r) {
        const int m = m0 + r, col = tn * 16 + n, grow = r0 + m;
        float cv = tanhf(c[r]);
        float hv = hw[m * 64 + col];
        float z  = rzw[m * 128 + 64 + col];
        float mk = mask[(size_t)grow * T_SZ + t];
        float hn = z * hv + (1.f - z) * cv;
        hn = mk * hn + (1.f - mk) * hv;
        hw[m * 64 + col] = hn;
        if (seq_out) seq_out[((size_t)grow * T_SZ + t) * 64 + col] = (_Float16)hn;
      }
    }
    __syncthreads();
  }

  if (final_out)
    for (int i = lane; i < 16 * 64; i += 32)
      final_out[(size_t)(r0 + (i >> 6)) * 64 + (i & 63)] = hw[i];
}

// ---------------------------------------------------------------------------
// Kernel: DIN attention MLP per 16-row tile of flattened [B*T] rows.
// din features built directly in A-fragment registers (no LDS staging);
// Wa1 (40 KB f16) staged once per block into LDS with async-to-LDS copies,
// shared by all 4 waves -> 4x less L2 weight traffic on the dominant GEMM.
// ---------------------------------------------------------------------------
__global__ void __launch_bounds__(128) k_attention(
    const _Float16* __restrict__ rnn1, const _Float16* __restrict__ item_f16,
    const _Float16* __restrict__ Wa1h,                     // [256,80] f16
    const float* __restrict__ ba1,
    const _Float16* __restrict__ Wa2h,                     // [96,48] f16 zero-padded
    const float* __restrict__ ba2,
    const float* __restrict__ Wa3, const float* __restrict__ ba3,
    const float* __restrict__ mask, float* __restrict__ scores) {
  __shared__ _Float16 sWa1[256 * 80];   // 40 KB, block-shared B matrix
  __shared__ _Float16 d1[4][16][96];    // 12 KB
  __shared__ _Float16 d2[4][16][48];    //  6 KB
  const int wave = threadIdx.x >> 5, lane = threadIdx.x & 31;
  const int tid = threadIdx.x;
  const int r0 = (blockIdx.x * 4 + wave) * 16;

  // ---- stage Wa1 into LDS: 2560 x 16B transfers, ASYNCcnt-tracked ----
  {
    unsigned lds_base = (unsigned)(size_t)(void*)&sWa1[0];
#pragma unroll
    for (int it = 0; it < 20; ++it) {
      int idx = tid + it * 128;  // 16-byte chunk index, 2560 total
      unsigned ldsa = lds_base + (unsigned)idx * 16u;
      const _Float16* gp = Wa1h + (size_t)idx * 8;
      asm volatile("global_load_async_to_lds_b128 %0, %1, off"
                   :: "v"(ldsa), "v"(gp) : "memory");
    }
    asm volatile("s_wait_asynccnt 0x0" ::: "memory");
  }
  for (int i = lane; i < 16 * 16; i += 32) d1[wave][i >> 4][80 + (i & 15)] = (_Float16)0.f;
  __syncthreads();

  // ---- build the 8 A-fragments of din = [q | k | q-k | q*k] in registers ----
  const int m = lane & 15, hi = (lane >> 4) * 8;
  const int g = r0 + m, b = g / T_SZ;
  const _Float16* qrow = item_f16 + (size_t)b * 64;
  const _Float16* krow = rnn1 + (size_t)g * 64;
  v16h fq0, fq1, fk0, fk1, fd0, fd1, fm0, fm1;
#pragma unroll
  for (int j = 0; j < 8; ++j) {
    int k = ((j < 4) ? 0 : 16) + hi + 2 * (j & 3);
    _Float16 q0 = qrow[k], q1 = qrow[k + 1], q2 = qrow[k + 32], q3 = qrow[k + 33];
    _Float16 k0 = krow[k], k1 = krow[k + 1], k2 = krow[k + 32], k3 = krow[k + 33];
    fq0[2 * j] = q0;      fq0[2 * j + 1] = q1;
    fq1[2 * j] = q2;      fq1[2 * j + 1] = q3;
    fk0[2 * j] = k0;      fk0[2 * j + 1] = k1;
    fk1[2 * j] = k2;      fk1[2 * j + 1] = k3;
    fd0[2 * j] = q0 - k0; fd0[2 * j + 1] = q1 - k1;
    fd1[2 * j] = q2 - k2; fd1[2 * j + 1] = q3 - k3;
    fm0[2 * j] = q0 * k0; fm0[2 * j + 1] = q1 * k1;
    fm1[2 * j] = q2 * k2; fm1[2 * j + 1] = q3 * k3;
  }

  // ---- layer 1: sigmoid(din @ Wa1 + ba1), N = 80, K = 256 (B from LDS) ----
#pragma unroll
  for (int tn = 0; tn < 5; ++tn) {
    v8f c = bias_frag(ba1, tn * 16);
    c = WMMA_F16(fq0, load_b_f16(sWa1, 80, 0,   tn * 16), c);
    c = WMMA_F16(fq1, load_b_f16(sWa1, 80, 32,  tn * 16), c);
    c = WMMA_F16(fk0, load_b_f16(sWa1, 80, 64,  tn * 16), c);
    c = WMMA_F16(fk1, load_b_f16(sWa1, 80, 96,  tn * 16), c);
    c = WMMA_F16(fd0, load_b_f16(sWa1, 80, 128, tn * 16), c);
    c = WMMA_F16(fd1, load_b_f16(sWa1, 80, 160, tn * 16), c);
    c = WMMA_F16(fm0, load_b_f16(sWa1, 80, 192, tn * 16), c);
    c = WMMA_F16(fm1, load_b_f16(sWa1, 80, 224, tn * 16), c);
#pragma unroll
    for (int r = 0; r < 8; ++r) c[r] = sigm(c[r]);
    store_d_f16(&d1[wave][0][0], 96, tn * 16, c);
  }
  __syncthreads();

  // ---- layer 2: sigmoid(d1 @ Wa2 + ba2), N = 48 (40 valid), K = 96 ----
#pragma unroll
  for (int tn = 0; tn < 3; ++tn) {
    v8f c = bias_frag_lim(ba2, tn * 16, 40);
#pragma unroll
    for (int kk = 0; kk < 3; ++kk)
      c = WMMA_F16(load_a_f16(&d1[wave][0][0], 96, kk * 32), load_b_f16(Wa2h, 48, kk * 32, tn * 16), c);
#pragma unroll
    for (int r = 0; r < 8; ++r) c[r] = sigm(c[r]);
    store_d_f16(&d2[wave][0][0], 48, tn * 16, c);
  }
  __syncthreads();

  // ---- layer 3: score = d2 . Wa3 + ba3, masked ----
  if (lane < 16) {
    const int gg = r0 + lane;
    float s = ba3[0];
#pragma unroll
    for (int j = 0; j < 40; ++j) s += (float)d2[wave][lane][j] * Wa3[j];
    scores[gg] = (mask[gg] > 0.f) ? s : -1e9f;
  }
}

// ---------------------------------------------------------------------------
// Kernel: masked softmax over T per batch row (one wave per row)
// ---------------------------------------------------------------------------
__global__ void k_softmax(const float* __restrict__ sc, float* __restrict__ al) {
  const int wave = threadIdx.x >> 5, lane = threadIdx.x & 31;
  const int row = blockIdx.x * 4 + wave;
  const float* s = sc + (size_t)row * T_SZ;
  float* a = al + (size_t)row * T_SZ;
  float mx = -3.4e38f;
  for (int t = lane; t < T_SZ; t += 32) mx = fmaxf(mx, s[t]);
#pragma unroll
  for (int o = 16; o > 0; o >>= 1) mx = fmaxf(mx, __shfl_xor(mx, o, 32));
  float sum = 0.f;
  for (int t = lane; t < T_SZ; t += 32) { float e = __expf(s[t] - mx); a[t] = e; sum += e; }
#pragma unroll
  for (int o = 16; o > 0; o >>= 1) sum += __shfl_xor(sum, o, 32);
  float inv = 1.f / sum;
  for (int t = lane; t < T_SZ; t += 32) a[t] *= inv;
}

// ---------------------------------------------------------------------------
// Kernel: att_in = rnn1 * alpha  (elementwise, f16 out)
// ---------------------------------------------------------------------------
__global__ void k_scale(const _Float16* __restrict__ rnn1, const float* __restrict__ al,
                        _Float16* __restrict__ att, size_t n) {
  for (size_t i = (size_t)blockIdx.x * blockDim.x + threadIdx.x; i < n;
       i += (size_t)gridDim.x * blockDim.x)
    att[i] = (_Float16)((float)rnn1[i] * al[i >> 6]);
}

// ---------------------------------------------------------------------------
// Kernel: FCN head: [B,320] -> prelu(128) -> prelu(40, padded 48) -> logits
// ---------------------------------------------------------------------------
__global__ void __launch_bounds__(128) k_fcn(
    const float* __restrict__ user_f32, const float* __restrict__ item_f32,
    const float* __restrict__ hsum, const float* __restrict__ final2,
    const _Float16* __restrict__ W1h,                      // [320,128]
    const float* __restrict__ b1, const float* __restrict__ p1,
    const _Float16* __restrict__ W2h,                      // [128,48] padded
    const float* __restrict__ b2, const float* __restrict__ p2,
    const float* __restrict__ W3, const float* __restrict__ b3,
    float* __restrict__ out) {
  __shared__ _Float16 feat[4][16][320];  // 40 KB
  __shared__ _Float16 h1[4][16][128];    // 16 KB
  __shared__ _Float16 h2[4][16][48];     //  6 KB
  const int wave = threadIdx.x >> 5, lane = threadIdx.x & 31;
  const int r0 = (blockIdx.x * 4 + wave) * 16;
  const int n = lane & 15;

  for (int i = lane; i < 16 * 64; i += 32) {
    const int row = i >> 6, e = i & 63, g = r0 + row;
    float uv = user_f32[(size_t)g * 64 + e];
    float iv = item_f32[(size_t)g * 64 + e];
    float hv = hsum[(size_t)g * 64 + e];
    float fv = final2[(size_t)g * 64 + e];
    feat[wave][row][e]       = (_Float16)uv;
    feat[wave][row][64 + e]  = (_Float16)iv;
    feat[wave][row][128 + e] = (_Float16)hv;
    feat[wave][row][192 + e] = (_Float16)(iv * hv);
    feat[wave][row][256 + e] = (_Float16)fv;
  }
  __syncthreads();

#pragma unroll
  for (int tn = 0; tn < 8; ++tn) {  // N = 128
    v8f c = bias_frag(b1, tn * 16);
#pragma unroll
    for (int kk = 0; kk < 10; ++kk)  // K = 320
      c = WMMA_F16(load_a_f16(&feat[wave][0][0], 320, kk * 32), load_b_f16(W1h, 128, kk * 32, tn * 16), c);
    const float a = p1[tn * 16 + n];
#pragma unroll
    for (int r = 0; r < 8; ++r) c[r] = (c[r] >= 0.f) ? c[r] : a * c[r];
    store_d_f16(&h1[wave][0][0], 128, tn * 16, c);
  }
  __syncthreads();

#pragma unroll
  for (int tn = 0; tn < 3; ++tn) {  // N = 48 (40 valid)
    v8f c = bias_frag_lim(b2, tn * 16, 40);
#pragma unroll
    for (int kk = 0; kk < 4; ++kk)  // K = 128
      c = WMMA_F16(load_a_f16(&h1[wave][0][0], 128, kk * 32), load_b_f16(W2h, 48, kk * 32, tn * 16), c);
    const int col = tn * 16 + n;
    const float a = (col < 40) ? p2[col] : 0.f;
#pragma unroll
    for (int r = 0; r < 8; ++r) c[r] = (c[r] >= 0.f) ? c[r] : a * c[r];
    store_d_f16(&h2[wave][0][0], 48, tn * 16, c);
  }
  __syncthreads();

  if (lane < 16) {
    float s = b3[0];
#pragma unroll
    for (int j = 0; j < 40; ++j) s += (float)h2[wave][lane][j] * W3[j];
    out[r0 + lane] = s;
  }
}

// ---------------------------------------------------------------------------
// Host launch
// ---------------------------------------------------------------------------
extern "C" void kernel_launch(void* const* d_in, const int* in_sizes, int n_in,
                              void* d_out, int out_size, void* d_ws, size_t ws_size,
                              hipStream_t stream) {
  const int*   u      = (const int*)d_in[0];
  const int*   iid    = (const int*)d_in[1];
  const int*   hist   = (const int*)d_in[2];
  const float* mask   = (const float*)d_in[3];
  const float* item_w = (const float*)d_in[4];
  const float* user_w = (const float*)d_in[5];
  const float* g1Wg = (const float*)d_in[6],  *g1Ug = (const float*)d_in[7];
  const float* g1bg = (const float*)d_in[8],  *g1Wc = (const float*)d_in[9];
  const float* g1Uc = (const float*)d_in[10], *g1bc = (const float*)d_in[11];
  const float* Wa1  = (const float*)d_in[12], *ba1  = (const float*)d_in[13];
  const float* Wa2  = (const float*)d_in[14], *ba2  = (const float*)d_in[15];
  const float* Wa3  = (const float*)d_in[16], *ba3  = (const float*)d_in[17];
  const float* g2Wg = (const float*)d_in[18], *g2Ug = (const float*)d_in[19];
  const float* g2bg = (const float*)d_in[20], *g2Wc = (const float*)d_in[21];
  const float* g2Uc = (const float*)d_in[22], *g2bc = (const float*)d_in[23];
  const float* W1 = (const float*)d_in[24], *b1 = (const float*)d_in[25], *p1 = (const float*)d_in[26];
  const float* W2 = (const float*)d_in[27], *b2 = (const float*)d_in[28], *p2 = (const float*)d_in[29];
  const float* W3 = (const float*)d_in[30], *b3 = (const float*)d_in[31];
  float* logits = (float*)d_out;

  // workspace carving (256-byte aligned)
  size_t off = 0;
  auto carve = [&](size_t bytes) -> void* {
    off = (off + 255) & ~(size_t)255;
    void* p = (char*)d_ws + off;
    off += bytes;
    return p;
  };
  const size_t seq_elems = (size_t)B_SZ * T_SZ * 64;
  _Float16* hist_f16 = (_Float16*)carve(seq_elems * 2);
  _Float16* rnn1_f16 = (_Float16*)carve(seq_elems * 2);
  _Float16* att_f16  = (_Float16*)carve(seq_elems * 2);
  _Float16* item_f16 = (_Float16*)carve((size_t)B_SZ * 64 * 2);
  float* item_f32 = (float*)carve((size_t)B_SZ * 64 * 4);
  float* user_f32 = (float*)carve((size_t)B_SZ * 64 * 4);
  float* hsum_f32 = (float*)carve((size_t)B_SZ * 64 * 4);
  float* fin2_f32 = (float*)carve((size_t)B_SZ * 64 * 4);
  float* scores   = (float*)carve((size_t)B_SZ * T_SZ * 4);
  float* alpha    = (float*)carve((size_t)B_SZ * T_SZ * 4);
  _Float16* hWg1 = (_Float16*)carve(64 * 128 * 2);
  _Float16* hUg1 = (_Float16*)carve(64 * 128 * 2);
  _Float16* hWc1 = (_Float16*)carve(64 * 64 * 2);
  _Float16* hUc1 = (_Float16*)carve(64 * 64 * 2);
  _Float16* hWa1 = (_Float16*)carve(256 * 80 * 2);
  _Float16* hWa2 = (_Float16*)carve(96 * 48 * 2);   // zero-padded
  _Float16* hWg2 = (_Float16*)carve(64 * 128 * 2);
  _Float16* hUg2 = (_Float16*)carve(64 * 128 * 2);
  _Float16* hWc2 = (_Float16*)carve(64 * 64 * 2);
  _Float16* hUc2 = (_Float16*)carve(64 * 64 * 2);
  _Float16* hW1  = (_Float16*)carve(320 * 128 * 2);
  _Float16* hW2  = (_Float16*)carve(128 * 48 * 2);  // zero-padded
  (void)ws_size; (void)in_sizes; (void)n_in; (void)out_size;

  auto cvt = [&](const float* s, _Float16* d, int R, int C, int R2, int C2) {
    int nel = R2 * C2;
    k_cvt_pad<<<(nel + 255) / 256, 256, 0, stream>>>(s, d, R, C, R2, C2);
  };
  cvt(g1Wg, hWg1, 64, 128, 64, 128);
  cvt(g1Ug, hUg1, 64, 128, 64, 128);
  cvt(g1Wc, hWc1, 64, 64, 64, 64);
  cvt(g1Uc, hUc1, 64, 64, 64, 64);
  cvt(Wa1,  hWa1, 256, 80, 256, 80);
  cvt(Wa2,  hWa2, 80, 40, 96, 48);
  cvt(g2Wg, hWg2, 64, 128, 64, 128);
  cvt(g2Ug, hUg2, 64, 128, 64, 128);
  cvt(g2Wc, hWc2, 64, 64, 64, 64);
  cvt(g2Uc, hUc2, 64, 64, 64, 64);
  cvt(W1,   hW1, 320, 128, 320, 128);
  cvt(W2,   hW2, 128, 40, 128, 48);

  k_gather<<<B_SZ, 256, 0, stream>>>(u, iid, hist, item_w, user_w,
                                     hist_f16, hsum_f32, item_f32, item_f16, user_f32);

  // GRU1: write per-step states
  k_gru_scan<<<B_SZ / 64, 128, 0, stream>>>(hist_f16, hWg1, hUg1, hWc1, hUc1,
                                            g1bg, g1bc, mask, rnn1_f16, nullptr);

  k_attention<<<(B_SZ * T_SZ) / 64, 128, 0, stream>>>(rnn1_f16, item_f16, hWa1, ba1,
                                                      hWa2, ba2, Wa3, ba3, mask, scores);

  k_softmax<<<B_SZ / 4, 128, 0, stream>>>(scores, alpha);

  k_scale<<<2048, 256, 0, stream>>>(rnn1_f16, alpha, att_f16, seq_elems);

  // GRU2: only final state
  k_gru_scan<<<B_SZ / 64, 128, 0, stream>>>(att_f16, hWg2, hUg2, hWc2, hUc2,
                                            g2bg, g2bc, mask, nullptr, fin2_f32);

  k_fcn<<<B_SZ / 64, 128, 0, stream>>>(user_f32, item_f32, hsum_f32, fin2_f32,
                                       hW1, b1, p1, hW2, b2, p2, W3, b3, logits);
}
